// BlockwiseQuantLinear_6244882448499
// MI455X (gfx1250) — 8-bit matrix operand order repaired and checked by reading, compiled, not run on hardware
//
#include <hip/hip_runtime.h>
#include <hip/hip_bf16.h>

typedef __attribute__((ext_vector_type(16))) int   v16i;
typedef __attribute__((ext_vector_type(8)))  float v8f;
typedef __attribute__((ext_vector_type(4)))  int   v4i;

#define MDIM 8192
#define KDIM 4096
#define NDIM 4096
#define KB   (KDIM / 128)
#define FP8_MAX 448.0f

#define LDS_STRIDE 144

#if __has_builtin(__builtin_amdgcn_global_load_async_to_lds_b128) && \
    __has_builtin(__builtin_amdgcn_s_wait_asynccnt)
#define HAVE_ASYNC_LDS 1
#define ASYNC_COPY16(g, l)                                              \
  __builtin_amdgcn_global_load_async_to_lds_b128(                       \
      (__attribute__((address_space(1))) v4i*)(g),                      \
      (__attribute__((address_space(3))) v4i*)(l), 0, 0)
#define WAIT_ASYNC(n) __builtin_amdgcn_s_wait_asynccnt(n)
#else
#define HAVE_ASYNC_LDS 0
#define ASYNC_COPY16(g, l) (*(int4*)(l) = *(const int4*)(g))
#define WAIT_ASYNC(n) ((void)0)
#endif


__device__ inline float bf16_f(unsigned short h) {
  union { unsigned int u; float f; } c;
  c.u = ((unsigned int)h) << 16;
  return c.f;
}

__device__ inline unsigned int fp8e4m3_byte(float f) {
  unsigned int u = __float_as_uint(f);
  unsigned int sign = (u >> 24) & 0x80u;
  unsigned int a = u & 0x7FFFFFFFu;
  if (a >= 0x43E00000u) return sign | 0x7Eu;
  if (a < 0x3C000000u) {
    float m = __uint_as_float(a) * 512.0f;
    int q = (int)(m + 0.5f);
    return sign | (unsigned int)(q > 7 ? 8 : q);
  }
  unsigned int lsb = (a >> 20) & 1u;
  a += 0x7FFFFu + lsb;
  unsigned int mant = (a >> 20) & 7u;
  int e = (int)(a >> 23) - 120;
  if (e >= 16 || (e == 15 && mant == 7)) return sign | 0x7Eu;
  if (e < 1) e = 1;
  return sign | ((unsigned int)e << 3) | mant;
}


__global__ __launch_bounds__(256) void quant_kernel(
    const unsigned short* __restrict__ x,
    unsigned char* __restrict__ xq,
    float* __restrict__ a_s) {
  const int lane = threadIdx.x & 31;
  const int wave = threadIdx.x >> 5;
  const long long task = (long long)blockIdx.x * 8 + wave;
  const int m  = (int)(task >> 5);
  const int kb = (int)(task & 31);
  const long long base = (long long)m * KDIM + kb * 128 + lane * 4;

  const ushort2* p = (const ushort2*)(x + base);
  ushort2 h01 = p[0];
  ushort2 h23 = p[1];
  float f0 = bf16_f(h01.x), f1 = bf16_f(h01.y);
  float f2 = bf16_f(h23.x), f3 = bf16_f(h23.y);

  float amax = fmaxf(fmaxf(fabsf(f0), fabsf(f1)), fmaxf(fabsf(f2), fabsf(f3)));
  #pragma unroll
  for (int off = 16; off > 0; off >>= 1)
    amax = fmaxf(amax, __shfl_xor(amax, off, 32));

  const float scale = fmaxf(amax, 1e-4f) / FP8_MAX;
  const float inv   = FP8_MAX / fmaxf(amax, 1e-4f);
  float q0 = f0 * inv, q1 = f1 * inv, q2 = f2 * inv, q3 = f3 * inv;

  unsigned int packed;
#if __has_builtin(__builtin_amdgcn_cvt_pk_fp8_f32)
  packed = (unsigned int)__builtin_amdgcn_cvt_pk_fp8_f32(q0, q1, 0, false);
  packed = (unsigned int)__builtin_amdgcn_cvt_pk_fp8_f32(q2, q3, (int)packed, true);
#else
  packed = fp8e4m3_byte(q0) | (fp8e4m3_byte(q1) << 8) |
           (fp8e4m3_byte(q2) << 16) | (fp8e4m3_byte(q3) << 24);
#endif
  *(unsigned int*)(xq + base) = packed;
  if (lane == 0) a_s[(long long)m * KB + kb] = scale;
}


__device__ inline void stage_tiles(const unsigned char* __restrict__ xq,
                                   const unsigned char* __restrict__ w,
                                   unsigned char* ldsAbuf, unsigned char* ldsBbuf,
                                   int m_wg, int n_wg, int k0, int tid) {
  #pragma unroll
  for (int i = 0; i < 2; ++i) {
    const int c = tid + i * 256;
    const int row = c >> 3, off = (c & 7) * 16;
    const unsigned char* g = xq + (long long)(m_wg + row) * KDIM + k0 + off;
    unsigned char* l = ldsAbuf + row * LDS_STRIDE + off;
    ASYNC_COPY16(g, l);
  }
  #pragma unroll
  for (int i = 0; i < 4; ++i) {
    const int c = tid + i * 256;
    const int row = c >> 3, off = (c & 7) * 16;
    const unsigned char* g = w + (long long)(n_wg + row) * KDIM + k0 + off;
    unsigned char* l = ldsBbuf + row * LDS_STRIDE + off;
    ASYNC_COPY16(g, l);
  }
}

__global__ __launch_bounds__(256) void gemm_kernel(
    const unsigned char* __restrict__ xq,
    const float* __restrict__ a_s,
    const unsigned char* __restrict__ w,
    const float* __restrict__ w_scale,
    float* __restrict__ out) {
  __shared__ unsigned char ldsA[2][64  * LDS_STRIDE];
  __shared__ unsigned char ldsB[2][128 * LDS_STRIDE];
  __shared__ float lds_as[64 * 33];

  const int tid    = threadIdx.x;
  const int lane   = tid & 31;
  const int wave   = tid >> 5;
  const int wave_n = wave & 3;
  const int wave_m = wave >> 2;
  const int nb     = blockIdx.x;
  const int m_wg   = blockIdx.y * 64;
  const int n_wg   = nb * 128;

  {
    const int r  = tid >> 2;
    const int c0 = (tid & 3) * 8;
    const float* src = a_s + (long long)(m_wg + r) * KB + c0;
    float* dst = lds_as + r * 33 + c0;
    #pragma unroll
    for (int i = 0; i < 8; ++i) dst[i] = src[i];
  }

  const int lhalf = lane >> 4;
  const int l15   = lane & 15;

  v8f acc[2][2];
  #pragma unroll
  for (int tm = 0; tm < 2; ++tm)
    #pragma unroll
    for (int tn = 0; tn < 2; ++tn)
      acc[tm][tn] = (v8f){};

  const float* wsrow = w_scale + (long long)nb * KB;

  stage_tiles(xq, w, ldsA[0], ldsB[0], m_wg, n_wg, 0, tid);

  for (int kb = 0; kb < KB; ++kb) {
    const int b = kb & 1;
    if (kb + 1 < KB) {
      stage_tiles(xq, w, ldsA[b ^ 1], ldsB[b ^ 1], m_wg, n_wg, (kb + 1) * 128, tid);
      WAIT_ASYNC(6);
    } else {
      WAIT_ASYNC(0);
    }
    __syncthreads();

    const float ws = wsrow[kb];

    v16i afrag[2];
    #pragma unroll
    for (int tm = 0; tm < 2; ++tm) {
      const unsigned char* ap =
          &ldsA[b][(wave_m * 32 + tm * 16 + l15) * LDS_STRIDE + 16 * lhalf];
      #pragma unroll
      for (int pp = 0; pp < 8; ++pp) {
        int2 v = *(const int2*)(ap + 32 * (pp >> 1) + 8 * (pp & 1));
        afrag[tm][2 * pp]     = v.x;
        afrag[tm][2 * pp + 1] = v.y;
      }
    }

    v16i bfrag[2];
    #pragma unroll
    for (int tn = 0; tn < 2; ++tn) {
      const unsigned char* bp =
          &ldsB[b][(wave_n * 32 + tn * 16 + l15) * LDS_STRIDE + 16 * lhalf];
      #pragma unroll
      for (int g = 0; g < 4; ++g) {
        int4 v = *(const int4*)(bp + 32 * g);
        bfrag[tn][4 * g]     = v.x;
        bfrag[tn][4 * g + 1] = v.y;
        bfrag[tn][4 * g + 2] = v.z;
        bfrag[tn][4 * g + 3] = v.w;
      }
    }

    #pragma unroll
    for (int tm = 0; tm < 2; ++tm) {
      const int mrel = wave_m * 32 + tm * 16 + 8 * lhalf;
      float s[8];
      #pragma unroll
      for (int r = 0; r < 8; ++r)
        s[r] = ws * lds_as[(mrel + r) * 33 + kb];

      #pragma unroll
      for (int tn = 0; tn < 2; ++tn) {
        v8f zero = (v8f){};
        v8f part = __builtin_amdgcn_wmma_f32_16x16x128_fp8_fp8(
            afrag[tm], bfrag[tn], (short)0, zero, false, false);
        #pragma unroll
        for (int r = 0; r < 8; ++r)
          acc[tm][tn][r] += s[r] * part[r];
      }
    }

    __syncthreads();
  }

  #pragma unroll
  for (int tm = 0; tm < 2; ++tm) {
    const int rowbase = m_wg + wave_m * 32 + tm * 16 + 8 * lhalf;
    #pragma unroll
    for (int tn = 0; tn < 2; ++tn) {
      const int col = n_wg + wave_n * 32 + tn * 16 + l15;
      #pragma unroll
      for (int r = 0; r < 8; ++r)
        out[(long long)(rowbase + r) * NDIM + col] = acc[tm][tn][r];
    }
  }
}


extern "C" void kernel_launch(void* const* d_in, const int* in_sizes, int n_in,
                              void* d_out, int out_size, void* d_ws, size_t ws_size,
                              hipStream_t stream) {
  const unsigned short* x      = (const unsigned short*)d_in[0];
  const unsigned char*  weight = (const unsigned char*)d_in[1];
  const float*          wscale = (const float*)d_in[2];
  float* out = (float*)d_out;

  unsigned char* xq  = (unsigned char*)d_ws;
  float*         a_s = (float*)((char*)d_ws + (size_t)MDIM * KDIM);

  quant_kernel<<<dim3(MDIM * KB / 8), dim3(256), 0, stream>>>(x, xq, a_s);

  gemm_kernel<<<dim3(NDIM / 128, MDIM / 64), dim3(256), 0, stream>>>(
      xq, a_s, weight, wscale, out);
}
